// DualBlockMultiRes_8804682957477
// MI455X (gfx1250) — matrix operand order repaired and checked by reading, not compiled, not run on hardware
//
#include <hip/hip_runtime.h>
#include <hip/hip_bf16.h>

// ---------------------------------------------------------------------------
// DualBlockMultiRes on MI455X (gfx1250), wave32 + v_wmma_f32_16x16x32_bf16.
// fp32 in memory, bf16 operands via v_cvt_pk_bf16_f32, fp32 accumulation;
// 128-bit global loads for fragments; DPP16 row_ror reductions for softmax.
// ---------------------------------------------------------------------------

#define BB 2
#define NN 2048
#define CC 768
#define QNN 1024
#define QCC 512
#define LNN 512
#define LCC 64
#define HH 12
#define DHH 64
#define QHH 8
#define QDHH 64

typedef __attribute__((ext_vector_type(16))) __bf16 v16bf;
typedef __attribute__((ext_vector_type(2)))  __bf16 v2bf;
typedef __attribute__((ext_vector_type(8)))  float  v8f;

union V16 {               // assemble a WMMA 16-bf16 fragment from packed pairs
    v16bf v;
    v2bf  p[8];
};

__device__ __forceinline__ __bf16 f2bf(float f) {
    return (__bf16)f;                    // native fptrunc f32 -> bf16
}

__device__ __forceinline__ v2bf f2bf2(float lo, float hi) {
    v2bf out;
    out[0] = (__bf16)lo;                 // fused into one v_cvt_pk_bf16_f32
    out[1] = (__bf16)hi;
    return out;
}

__device__ __forceinline__ v8f wmma_bf16(v16bf a, v16bf b, v8f c) {
    return __builtin_amdgcn_wmma_f32_16x16x32_bf16(
        false, a, false, b, (short)0, c, false, false);
}

// A-fragment K index (contiguous order k = 16*half + s): VGPR v holds K pair.
__device__ __forceinline__ int a_kidx(int v, int half) {
    return half * 16 + 2 * v;
}

// Build the 8 packed pairs of an A-style fragment from two contiguous 8-float
// runs: src[half*16 .. +8) -> p[0..3], src[half*16+8 .. +8) -> p[4..7].
__device__ __forceinline__ void load_afrag(const float* __restrict__ src,
                                           int half, float scale, V16& f) {
    const float4* lo = (const float4*)(src + half * 16);
    const float4* hi = (const float4*)(src + half * 16 + 8);
    float4 a0 = lo[0], a1 = lo[1], a2 = hi[0], a3 = hi[1];
    f.p[0] = f2bf2(a0.x * scale, a0.y * scale);
    f.p[1] = f2bf2(a0.z * scale, a0.w * scale);
    f.p[2] = f2bf2(a1.x * scale, a1.y * scale);
    f.p[3] = f2bf2(a1.z * scale, a1.w * scale);
    f.p[4] = f2bf2(a2.x * scale, a2.y * scale);
    f.p[5] = f2bf2(a2.z * scale, a2.w * scale);
    f.p[6] = f2bf2(a3.x * scale, a3.y * scale);
    f.p[7] = f2bf2(a3.z * scale, a3.w * scale);
}

// Build the 8 packed pairs of a B-style fragment from one contiguous 16-float
// run: src[half*16 .. +16) -> p[0..7].
__device__ __forceinline__ void load_bfrag(const float* __restrict__ src,
                                           int half, V16& f) {
    const float4* q = (const float4*)(src + half * 16);
    float4 b0 = q[0], b1 = q[1], b2 = q[2], b3 = q[3];
    f.p[0] = f2bf2(b0.x, b0.y);  f.p[1] = f2bf2(b0.z, b0.w);
    f.p[2] = f2bf2(b1.x, b1.y);  f.p[3] = f2bf2(b1.z, b1.w);
    f.p[4] = f2bf2(b2.x, b2.y);  f.p[5] = f2bf2(b2.z, b2.w);
    f.p[6] = f2bf2(b3.x, b3.y);  f.p[7] = f2bf2(b3.z, b3.w);
}

// ---------------------------------------------------------------------------
// 16-lane row reductions via DPP16 row_ror (pure VALU, no LDS traffic).
// Rotate-by-k log reduction: after k = 1,2,4,8 every lane in a 16-lane DPP
// row holds the full reduction; both wave32 halves reduce independently,
// which matches the WMMA D-layout half-groups.
// ---------------------------------------------------------------------------
template <int CTRL>
__device__ __forceinline__ float dpp_rot(float x) {
    int xi = __builtin_bit_cast(int, x);
    int r  = __builtin_amdgcn_update_dpp(xi, xi, CTRL, 0xF, 0xF, false);
    return __builtin_bit_cast(float, r);
}

__device__ __forceinline__ float rowmax16(float x) {
    x = fmaxf(x, dpp_rot<0x121>(x));   // row_ror:1
    x = fmaxf(x, dpp_rot<0x122>(x));   // row_ror:2
    x = fmaxf(x, dpp_rot<0x124>(x));   // row_ror:4
    x = fmaxf(x, dpp_rot<0x128>(x));   // row_ror:8
    return x;
}
__device__ __forceinline__ float rowsum16(float x) {
    x += dpp_rot<0x121>(x);
    x += dpp_rot<0x122>(x);
    x += dpp_rot<0x124>(x);
    x += dpp_rot<0x128>(x);
    return x;
}

// ---------------------------------------------------------------------------
// GEMM: C[M,N] = A[M,K] * W[N,K]^T (+ bias[N]).  Each wave: 16 rows x 32 cols
// (A fragment reused across two WMMAs); block (8 waves) = 16 x 256.
// Requires M % 16 == 0, N % 32 == 0, K % 32 == 0.
// ---------------------------------------------------------------------------
template <bool HAS_BIAS>
__global__ __launch_bounds__(256) void gemm_bf16_wmma(
    const float* __restrict__ A, const float* __restrict__ W,
    const float* __restrict__ bias, float* __restrict__ C,
    int M, int N, int K)
{
    const int wave = threadIdx.x >> 5;
    const int lane = threadIdx.x & 31;
    const int half = lane >> 4;
    const int lr   = lane & 15;
    const int m0 = blockIdx.y * 16;
    const int n0 = blockIdx.x * 256 + wave * 32;
    if (n0 >= N) return;                       // wave-uniform

    const float* __restrict__ arow = A + (size_t)(m0 + lr) * K;
    const float* __restrict__ w0   = W + (size_t)(n0 + lr) * K;
    const float* __restrict__ w1   = W + (size_t)(n0 + 16 + lr) * K;

    v8f acc0 = {}, acc1 = {};
    for (int k0 = 0; k0 < K; k0 += 32) {
        V16 a, b0, b1;
        load_afrag(arow + k0, half, 1.0f, a);
        load_bfrag(w0 + k0, half, b0);
        load_bfrag(w1 + k0, half, b1);
        acc0 = wmma_bf16(a.v, b0.v, acc0);
        acc1 = wmma_bf16(a.v, b1.v, acc1);
    }

    float bn0 = 0.0f, bn1 = 0.0f;
    if (HAS_BIAS) {
        bn0 = bias[n0 + lr];
        bn1 = bias[n0 + 16 + lr];
    }
#pragma unroll
    for (int v = 0; v < 8; ++v) {
        const int m = m0 + v + 8 * half;       // D: lane = column, rows in VGPRs
        C[(size_t)m * N + (n0 + lr)]      = acc0[v] + bn0;
        C[(size_t)m * N + (n0 + 16 + lr)] = acc1[v] + bn1;
    }
}

// ---------------------------------------------------------------------------
// Flash-attention inner loop over one KV span for one 16-row query tile.
// K rows: Kb + key*rs ; V rows: Vb + key*rs ; head_dim = 64, scale in Q.
// ---------------------------------------------------------------------------
__device__ void flash_span(const float* __restrict__ Kb,
                           const float* __restrict__ Vb,
                           long rs, int nkeys,
                           v16bf qf0, v16bf qf1,
                           v8f O[4], float mrow[8], float lrow[8],
                           __bf16* __restrict__ Pbuf, int lane)
{
    const int half = lane >> 4;
    const int lr   = lane & 15;

    for (int kb = 0; kb < nkeys; kb += 32) {
        // ---- S = Q * K^T for 32 keys (two 16x16 D tiles, K-depth 64) ----
        v8f S[2];
#pragma unroll
        for (int sub = 0; sub < 2; ++sub) {
            const float* __restrict__ krow =
                Kb + (size_t)(kb + sub * 16 + lr) * rs;
            V16 b0, b1;
            load_bfrag(krow, half, b0);        // d = 0..31
            load_bfrag(krow + 32, half, b1);   // d = 32..63
            v8f s = {};
            s = wmma_bf16(qf0, b0.v, s);
            s = wmma_bf16(qf1, b1.v, s);
            S[sub] = s;
        }

        // ---- online softmax (row stats live in a 16-lane half-group) ----
#pragma unroll
        for (int v = 0; v < 8; ++v) {
            float mx = rowmax16(fmaxf(S[0][v], S[1][v]));
            const float mnew = fmaxf(mrow[v], mx);
            const float corr = __expf(mrow[v] - mnew);
            const float p0 = __expf(S[0][v] - mnew);
            const float p1 = __expf(S[1][v] - mnew);
            lrow[v] = lrow[v] * corr + rowsum16(p0 + p1);
            mrow[v] = mnew;
#pragma unroll
            for (int t = 0; t < 4; ++t) O[t][v] *= corr;
            // stage P (16x32, bf16) for the D-layout -> A-layout transpose
            Pbuf[(v + 8 * half) * 32 + lr]      = f2bf(p0);
            Pbuf[(v + 8 * half) * 32 + 16 + lr] = f2bf(p1);
        }

        // per-wave LDS round trip; DS ops from one wave complete in order.
        asm volatile("" ::: "memory");
        V16 pa;
#pragma unroll
        for (int v = 0; v < 8; ++v) {
            const int kk = a_kidx(v, half);
            pa.p[v] = *(const v2bf*)(Pbuf + lr * 32 + kk);
        }

        // ---- O += P * V  (4 column tiles of 16 over head_dim 64) ----
#pragma unroll
        for (int t = 0; t < 4; ++t) {
            V16 bv;
#pragma unroll
            for (int v = 0; v < 8; ++v) {
                const float* __restrict__ vr =
                    Vb + (size_t)(kb + half * 16 + 2 * v) * rs + 16 * t + lr;
                bv.p[v] = f2bf2(vr[0], vr[rs]);
            }
            O[t] = wmma_bf16(pa.v, bv.v, O[t]);
        }
    }
}

__device__ __forceinline__ void load_qfrags(const float* __restrict__ qrow,
                                            int half, v16bf& qf0, v16bf& qf1)
{
    V16 f0, f1;
    load_afrag(qrow,      half, 0.125f, f0);   // 1/sqrt(64) folded into Q
    load_afrag(qrow + 32, half, 0.125f, f1);
    qf0 = f0.v;
    qf1 = f1.v;
}

// ---------------------------------------------------------------------------
// Self-attention over x: QKV buffer rows = tokens, cols [3][H][DH].
// grid = (N/128, B*H); each wave owns a 16-row query tile.
// ---------------------------------------------------------------------------
__global__ __launch_bounds__(256) void attn_x_kernel(
    const float* __restrict__ QKV, float* __restrict__ XO)
{
    __shared__ __bf16 Pb[8][16 * 32];
    const int wave = threadIdx.x >> 5;
    const int lane = threadIdx.x & 31;
    const int half = lane >> 4;
    const int lr   = lane & 15;
    const int b = blockIdx.y / HH;
    const int h = blockIdx.y % HH;
    const int q0 = blockIdx.x * 128 + wave * 16;

    const long rs = 3 * CC;
    const float* Qb = QKV + (size_t)b * NN * rs + h * DHH;
    const float* Kb = Qb + CC;
    const float* Vb = Qb + 2 * CC;

    v16bf qf0, qf1;
    load_qfrags(Qb + (size_t)(q0 + lr) * rs, half, qf0, qf1);

    v8f O[4] = {};
    float mrow[8], lrow[8];
#pragma unroll
    for (int v = 0; v < 8; ++v) { mrow[v] = -3.0e38f; lrow[v] = 0.0f; }

    flash_span(Kb, Vb, rs, NN, qf0, qf1, O, mrow, lrow, Pb[wave], lane);

#pragma unroll
    for (int t = 0; t < 4; ++t)
#pragma unroll
        for (int v = 0; v < 8; ++v) {
            const int m = q0 + v + 8 * half;
            XO[((size_t)b * NN + m) * CC + h * DHH + 16 * t + lr] =
                O[t][v] / lrow[v];
        }
}

// ---------------------------------------------------------------------------
// Gated cross-attention for the query stream:
//   out = tanh(g_h) * softmax(Q.Kx^T).Vx  +  softmax(Q.Kq^T).Vq
// ---------------------------------------------------------------------------
__global__ __launch_bounds__(256) void attn_q_kernel(
    const float* __restrict__ KV2, const float* __restrict__ QQKV,
    const float* __restrict__ gate, float* __restrict__ QO)
{
    __shared__ __bf16 Pb[8][16 * 32];
    const int wave = threadIdx.x >> 5;
    const int lane = threadIdx.x & 31;
    const int half = lane >> 4;
    const int lr   = lane & 15;
    const int b = blockIdx.y / QHH;
    const int h = blockIdx.y % QHH;
    const int q0 = blockIdx.x * 128 + wave * 16;

    const long rsq = 3 * QCC;
    const float* Qb = QQKV + (size_t)b * QNN * rsq + h * QDHH;

    v16bf qf0, qf1;
    load_qfrags(Qb + (size_t)(q0 + lr) * rsq, half, qf0, qf1);

    // phase 1: keys/values derived from x
    const long rs1 = 2 * QCC;
    const float* Kb1 = KV2 + (size_t)b * NN * rs1 + h * QDHH;
    const float* Vb1 = Kb1 + QCC;
    v8f O1[4] = {};
    float m1[8], l1[8];
#pragma unroll
    for (int v = 0; v < 8; ++v) { m1[v] = -3.0e38f; l1[v] = 0.0f; }
    flash_span(Kb1, Vb1, rs1, NN, qf0, qf1, O1, m1, l1, Pb[wave], lane);

    // phase 2: keys/values from the query stream itself
    const float* Kb2 = Qb + QCC;
    const float* Vb2 = Qb + 2 * QCC;
    v8f O2[4] = {};
    float m2[8], l2[8];
#pragma unroll
    for (int v = 0; v < 8; ++v) { m2[v] = -3.0e38f; l2[v] = 0.0f; }
    flash_span(Kb2, Vb2, rsq, QNN, qf0, qf1, O2, m2, l2, Pb[wave], lane);

    const float g = tanhf(gate[h]);
#pragma unroll
    for (int t = 0; t < 4; ++t)
#pragma unroll
        for (int v = 0; v < 8; ++v) {
            const int m = q0 + v + 8 * half;
            QO[((size_t)b * QNN + m) * QCC + h * QDHH + 16 * t + lr] =
                g * O1[t][v] / l1[v] + O2[t][v] / l2[v];
        }
}

// ---------------------------------------------------------------------------
// Reproduce the reference's low_res transpose(1,2).reshape scramble:
//   out[b, f] = low[b, f % LN, f / LN],  f in [0, LN*LC)
// ---------------------------------------------------------------------------
__global__ void scramble_low_kernel(const float* __restrict__ low,
                                    float* __restrict__ out)
{
    const int idx = blockIdx.x * blockDim.x + threadIdx.x;
    if (idx >= BB * LNN * LCC) return;
    const int b = idx / (LNN * LCC);
    const int r = idx % (LNN * LCC);
    out[idx] = low[(size_t)b * LNN * LCC + (r % LNN) * LCC + (r / LNN)];
}

// ---------------------------------------------------------------------------
extern "C" void kernel_launch(void* const* d_in, const int* in_sizes, int n_in,
                              void* d_out, int out_size, void* d_ws, size_t ws_size,
                              hipStream_t stream)
{
    const float* x       = (const float*)d_in[0];
    const float* query   = (const float*)d_in[1];
    const float* low_res = (const float*)d_in[2];
    const float* W_qkv   = (const float*)d_in[3];
    const float* W_xkv   = (const float*)d_in[4];
    const float* W_qlin  = (const float*)d_in[5];
    const float* gate    = (const float*)d_in[6];
    const float* W_proj  = (const float*)d_in[7];
    const float* b_proj  = (const float*)d_in[8];
    const float* W_qproj = (const float*)d_in[9];
    const float* b_qproj = (const float*)d_in[10];
    const float* W_lproj = (const float*)d_in[11];
    const float* b_lproj = (const float*)d_in[12];

    float* out = (float*)d_out;
    float* ws  = (float*)d_ws;

    // workspace layout (floats)
    float* QKV  = ws;                               // 4096 x 2304
    float* KV2  = QKV  + (size_t)4096 * 2304;       // 4096 x 1024
    float* QQKV = KV2  + (size_t)4096 * 1024;       // 2048 x 1536
    float* XO   = QQKV + (size_t)2048 * 1536;       // 4096 x 768
    float* QO   = XO   + (size_t)4096 * 768;        // 2048 x 512
    float* LS   = QO   + (size_t)2048 * 512;        // 1024 x 64

    const dim3 blk(256);

    // fused projections (no bias)
    gemm_bf16_wmma<false><<<dim3(9, 4096 / 16), blk, 0, stream>>>(
        x, W_qkv, nullptr, QKV, 4096, 2304, 768);
    gemm_bf16_wmma<false><<<dim3(4, 4096 / 16), blk, 0, stream>>>(
        x, W_xkv, nullptr, KV2, 4096, 1024, 768);
    gemm_bf16_wmma<false><<<dim3(6, 2048 / 16), blk, 0, stream>>>(
        query, W_qlin, nullptr, QQKV, 2048, 1536, 512);

    // attention
    attn_x_kernel<<<dim3(NN / 128, BB * HH), blk, 0, stream>>>(QKV, XO);
    attn_q_kernel<<<dim3(QNN / 128, BB * QHH), blk, 0, stream>>>(KV2, QQKV, gate, QO);

    // low_res scramble (reference layout bug reproduced)
    scramble_low_kernel<<<(BB * LNN * LCC + 255) / 256, blk, 0, stream>>>(low_res, LS);

    // output projections (+bias), writing straight into d_out
    float* out_x = out;
    float* out_q = out + (size_t)BB * NN * CC;
    float* out_l = out_q + (size_t)BB * QNN * QCC;
    gemm_bf16_wmma<true><<<dim3(3, 4096 / 16), blk, 0, stream>>>(
        XO, W_proj, b_proj, out_x, 4096, 768, 768);
    gemm_bf16_wmma<true><<<dim3(2, 2048 / 16), blk, 0, stream>>>(
        QO, W_qproj, b_qproj, out_q, 2048, 512, 512);
    gemm_bf16_wmma<true><<<dim3(1, 1024 / 16), blk, 0, stream>>>(
        LS, W_lproj, b_lproj, out_l, 1024, 64, 64);
}
